// TestConv_21474836480479
// MI455X (gfx1250) — compile-verified, matrix operand order checked by reading, not run on hardware
//
#include <hip/hip_runtime.h>
#include <hip/hip_bf16.h>

typedef __attribute__((ext_vector_type(16))) _Float16 v16h;
typedef __attribute__((ext_vector_type(8)))  float    v8f;
typedef __attribute__((ext_vector_type(4)))  unsigned int v4u;

union FragH { v4u q[2]; v16h h; };   // 32 bytes = 16 halfs = one WMMA A/B fragment per lane

#define IN_C 32
#define P_C  64
#define K_C  27
#define M_C  4
#define KC   (K_C * P_C)        // 1728 contraction length
#define KT_N (KC / 32)          // 54 k-tiles of 32
#define COLS (M_C * P_C)        // 256 output columns (m*64+d)
#define CT_N (COLS / 16)        // 16 column tiles
#define AROW 1736               // padded LDS row (1728 + 8 halfs) to stagger banks

// ---------------------------------------------------------------------------
// Kernel 1: x = relu(x_f@W_top+b), residual = x_f@W_down+b, v = relu(x@W_v+b)
// 256 threads = 4 voxels x 64 channels
// ---------------------------------------------------------------------------
__global__ __launch_bounds__(256) void prep_kernel(
    const float* __restrict__ x_f,
    const float* __restrict__ W_top, const float* __restrict__ b_top,
    const float* __restrict__ W_down, const float* __restrict__ b_down,
    const float* __restrict__ W_v, const float* __restrict__ b_v,
    float* __restrict__ x_out, float* __restrict__ res_out,
    _Float16* __restrict__ v_out, int N)
{
    __shared__ float xs[4][P_C];
    int t = threadIdx.x;
    int vloc = t >> 6, p = t & 63;
    int n = blockIdx.x * 4 + vloc;
    bool ok = n < N;
    int nn = ok ? n : 0;

    float accT = 0.f, accD = 0.f;
    #pragma unroll
    for (int c = 0; c < IN_C; ++c) {
        float xf = x_f[(size_t)nn * IN_C + c];
        accT += xf * W_top[c * P_C + p];
        accD += xf * W_down[c * P_C + p];
    }
    float xv = fmaxf(accT + b_top[p], 0.f);
    xs[vloc][p] = xv;
    __syncthreads();

    float accV = 0.f;
    #pragma unroll
    for (int c = 0; c < P_C; ++c)
        accV += xs[vloc][c] * W_v[c * P_C + p];

    if (ok) {
        x_out[(size_t)n * P_C + p]   = xv;
        res_out[(size_t)n * P_C + p] = accD + b_down[p];
        v_out[(size_t)n * P_C + p]   = (_Float16)fmaxf(accV + b_v[p], 0.f);
    }
}

// ---------------------------------------------------------------------------
// Kernel 2: q scalar (VD=1) + 4-way softmax attention. One wave32 per voxel.
// ---------------------------------------------------------------------------
__global__ __launch_bounds__(256) void attn_kernel(
    const float* __restrict__ x, const int* __restrict__ nbr_idx,
    const int* __restrict__ nbr_mask, const int* __restrict__ coords,
    const float* __restrict__ W_pos, const float* __restrict__ b_pos,
    const float* __restrict__ W_q, const float* __restrict__ b_q,
    const float* __restrict__ proj, float* __restrict__ attn_out, int N)
{
    int wave = threadIdx.x >> 5, lane = threadIdx.x & 31;
    int n = blockIdx.x * 8 + wave;
    if (n >= N) return;

    int c0 = lane, c1 = lane + 32;
    int cx = coords[n * 3 + 0], cy = coords[n * 3 + 1], cz = coords[n * 3 + 2];
    float acc = 0.f;
    for (int k = 0; k < K_C; ++k) {
        if (!nbr_mask[(size_t)n * K_C + k]) continue;
        int idx = nbr_idx[(size_t)n * K_C + k];
        float fx = (float)(coords[idx * 3 + 0] - cx);
        float fy = (float)(coords[idx * 3 + 1] - cy);
        float fz = (float)(coords[idx * 3 + 2] - cz);
        float p0 = fmaxf(fx * W_pos[c0] + fy * W_pos[P_C + c0] + fz * W_pos[2 * P_C + c0] + b_pos[c0], 0.f);
        float p1 = fmaxf(fx * W_pos[c1] + fy * W_pos[P_C + c1] + fz * W_pos[2 * P_C + c1] + b_pos[c1], 0.f);
        acc += (x[(size_t)idx * P_C + c0] + p0) * W_q[k * P_C + c0];
        acc += (x[(size_t)idx * P_C + c1] + p1) * W_q[k * P_C + c1];
    }
    // wave32 tree reduction
    for (int off = 16; off; off >>= 1) acc += __shfl_xor(acc, off, 32);

    if (lane == 0) {
        float q = fmaxf(acc + b_q[0], 0.f);
        float l[M_C], mx = -1e30f, s = 0.f;
        #pragma unroll
        for (int m = 0; m < M_C; ++m) { l[m] = q * proj[m] * 0.1f; mx = fmaxf(mx, l[m]); }
        #pragma unroll
        for (int m = 0; m < M_C; ++m) { l[m] = __expf(l[m] - mx); s += l[m]; }
        float inv = 1.f / s;
        #pragma unroll
        for (int m = 0; m < M_C; ++m) attn_out[(size_t)n * M_C + m] = l[m] * inv;
    }
}

// ---------------------------------------------------------------------------
// Kernel 3: pack W_code (f32) into per-lane f16 B fragments.
// Fragment (ct, kt): lane holds col = ct*16 + lane%16, kbase = (lane/16)*8,
// halfs 0..7 -> kc = kt*32+kbase+i, halfs 8..15 -> kc = kt*32+kbase+16+(i-8).
// ---------------------------------------------------------------------------
__global__ __launch_bounds__(256) void packB_kernel(
    const float* __restrict__ W_code, _Float16* __restrict__ Bfrag)
{
    int tid = blockIdx.x * 256 + threadIdx.x;
    if (tid >= CT_N * KT_N * 32) return;
    int lane = tid & 31;
    int kt = (tid >> 5) % KT_N;
    int ct = tid / (KT_N * 32);
    int kbase = (lane >= 16) ? 8 : 0;
    int col = ct * 16 + (lane & 15);
    int m = col >> 6, d = col & 63;

    _Float16* dst = Bfrag + (size_t)tid * 16;
    #pragma unroll
    for (int i = 0; i < 16; ++i) {
        int koff = (i < 8) ? (kbase + i) : (kbase + 16 + (i - 8));
        int kc = kt * 32 + koff;
        int k = kc >> 6, c = kc & 63;
        dst[i] = (_Float16)W_code[(((size_t)m * K_C + k) * P_C + c) * P_C + d];
    }
}

// ---------------------------------------------------------------------------
// Kernel 4: heavy codebook GEMM via WMMA + attention-weighted epilogue.
// Block = 256 threads (8 waves), owns 16 voxels. 54 k-tiles x 2 col-tiles/wave.
// Gather uses CDNA5 async global->LDS copies (per-lane gathered addresses),
// tracked by ASYNCcnt; masked neighbors are zero-filled with ds stores.
// ---------------------------------------------------------------------------
__global__ __launch_bounds__(256) void code_wmma_kernel(
    const _Float16* __restrict__ v_f16, const _Float16* __restrict__ Bfrag,
    const int* __restrict__ nbr_idx, const int* __restrict__ nbr_mask,
    const float* __restrict__ b_code, const float* __restrict__ attn,
    const float* __restrict__ residual, float* __restrict__ out, int N)
{
    // Alds declared FIRST -> sits at LDS segment offset 0 (async-copy LDS
    // destinations below are byte offsets relative to the LDS segment base).
    __shared__ _Float16 Alds[16 * AROW];
    __shared__ float accum[16 * P_C];

    int tid = threadIdx.x;
    int base = blockIdx.x * 16;

    for (int i = tid; i < 16 * P_C; i += 256) accum[i] = 0.f;

    // --- masked gather of nbr_v straight into the LDS A-panel -------------
    for (int chunk = tid; chunk < 16 * K_C * 8; chunk += 256) {
        int m   = chunk / (K_C * 8);
        int rem = chunk % (K_C * 8);
        int k   = rem >> 3;
        int c8  = rem & 7;
        int n   = base + m;
        unsigned lds_half = (unsigned)(m * AROW + k * P_C + c8 * 8);
        bool valid = false;
        size_t gidx = 0;
        if (n < N && nbr_mask[(size_t)n * K_C + k]) {
            valid = true;
            gidx = (size_t)nbr_idx[(size_t)n * K_C + k] * P_C + (size_t)(c8 * 8);
        }
        if (valid) {
            unsigned lds_byte = lds_half * (unsigned)sizeof(_Float16);
            unsigned long long gaddr = (unsigned long long)(const void*)(v_f16 + gidx);
            // CDNA5: async DMA 16B global -> LDS, per-lane addresses, ASYNCcnt
            asm volatile("global_load_async_to_lds_b128 %0, %1, off"
                         :: "v"(lds_byte), "v"(gaddr)
                         : "memory");
        } else {
            *reinterpret_cast<v4u*>(&Alds[lds_half]) = 0;   // zero row chunk
        }
    }
    asm volatile("s_wait_asynccnt 0" ::: "memory");
    __syncthreads();

    // --- WMMA main loop ----------------------------------------------------
    int wave = tid >> 5, lane = tid & 31;
    int t0 = wave * 2, t1 = wave * 2 + 1;
    int mrow = lane & 15;
    int kbase = (lane >> 4) * 8;

    v8f c0 = {};
    v8f c1 = {};
    const _Float16* B0 = Bfrag + (size_t)t0 * KT_N * 32 * 16;
    const _Float16* B1 = Bfrag + (size_t)t1 * KT_N * 32 * 16;

    for (int kt = 0; kt < KT_N; ++kt) {
        FragH a, b0, b1;
        const _Float16* ap = &Alds[mrow * AROW + kt * 32 + kbase];
        a.q[0] = *reinterpret_cast<const v4u*>(ap);
        a.q[1] = *reinterpret_cast<const v4u*>(ap + 16);

        const v4u* bp0 = reinterpret_cast<const v4u*>(B0 + (size_t)(kt * 32 + lane) * 16);
        b0.q[0] = bp0[0]; b0.q[1] = bp0[1];
        const v4u* bp1 = reinterpret_cast<const v4u*>(B1 + (size_t)(kt * 32 + lane) * 16);
        b1.q[0] = bp1[0]; b1.q[1] = bp1[1];

        if (kt + 1 < KT_N) {
            __builtin_prefetch((const void*)(B0 + (size_t)((kt + 1) * 32 + lane) * 16), 0, 3);
            __builtin_prefetch((const void*)(B1 + (size_t)((kt + 1) * 32 + lane) * 16), 0, 3);
        }

        c0 = __builtin_amdgcn_wmma_f32_16x16x32_f16(false, a.h, false, b0.h, (short)0, c0, false, false);
        c1 = __builtin_amdgcn_wmma_f32_16x16x32_f16(false, a.h, false, b1.h, (short)0, c1, false, false);
    }

    // --- epilogue: bias + ReLU, attention weight, LDS f32 atomic reduce ----
    #pragma unroll
    for (int which = 0; which < 2; ++which) {
        int t = which ? t1 : t0;
        int col = t * 16 + (lane & 15);
        int mcode = col >> 6, d = col & 63;
        float bc = b_code[mcode * P_C + d];
        #pragma unroll
        for (int r = 0; r < 8; ++r) {
            float cv = which ? c1[r] : c0[r];
            int vox = r + ((lane >> 4) << 3);
            int n = base + vox;
            float w = (n < N) ? attn[(size_t)n * M_C + mcode] : 0.f;
            float valf = fmaxf(cv + bc, 0.f);
            atomicAdd(&accum[vox * P_C + d], w * valf);   // ds_add_f32
        }
    }
    __syncthreads();

    // --- final: + residual, ReLU, store ------------------------------------
    for (int i = tid; i < 16 * P_C; i += 256) {
        int vox = i >> 6, d = i & 63;
        int n = base + vox;
        if (n < N)
            out[(size_t)n * P_C + d] = fmaxf(accum[i] + residual[(size_t)n * P_C + d], 0.f);
    }
}

// ---------------------------------------------------------------------------
extern "C" void kernel_launch(void* const* d_in, const int* in_sizes, int n_in,
                              void* d_out, int out_size, void* d_ws, size_t ws_size,
                              hipStream_t stream)
{
    const float* x_f      = (const float*)d_in[0];
    const int*   nbr_idx  = (const int*)d_in[1];
    const int*   nbr_mask = (const int*)d_in[2];
    const int*   coords   = (const int*)d_in[3];
    const float* W_top    = (const float*)d_in[4];
    const float* b_top    = (const float*)d_in[5];
    const float* W_down   = (const float*)d_in[6];
    const float* b_down   = (const float*)d_in[7];
    const float* W_pos    = (const float*)d_in[8];
    const float* b_pos    = (const float*)d_in[9];
    const float* W_q      = (const float*)d_in[10];
    const float* b_q      = (const float*)d_in[11];
    const float* proj     = (const float*)d_in[12];
    const float* W_v      = (const float*)d_in[13];
    const float* b_v      = (const float*)d_in[14];
    const float* W_code   = (const float*)d_in[15];
    const float* b_code   = (const float*)d_in[16];

    int N = in_sizes[0] / IN_C;

    auto align256 = [](size_t v) { return (v + 255) & ~(size_t)255; };
    char* ws = (char*)d_ws;
    size_t off = 0;
    float*    x_ws    = (float*)(ws + off);    off += align256((size_t)N * P_C * sizeof(float));
    float*    res_ws  = (float*)(ws + off);    off += align256((size_t)N * P_C * sizeof(float));
    _Float16* v_ws    = (_Float16*)(ws + off); off += align256((size_t)N * P_C * sizeof(_Float16));
    float*    attn_ws = (float*)(ws + off);    off += align256((size_t)N * M_C * sizeof(float));
    _Float16* Bfrag   = (_Float16*)(ws + off); off += align256((size_t)CT_N * KT_N * 32 * 16 * sizeof(_Float16));

    float* outp = (float*)d_out;

    prep_kernel<<<(N + 3) / 4, 256, 0, stream>>>(
        x_f, W_top, b_top, W_down, b_down, W_v, b_v, x_ws, res_ws, v_ws, N);

    attn_kernel<<<(N + 7) / 8, 256, 0, stream>>>(
        x_ws, nbr_idx, nbr_mask, coords, W_pos, b_pos, W_q, b_q, proj, attn_ws, N);

    packB_kernel<<<(CT_N * KT_N * 32 + 255) / 256, 256, 0, stream>>>(W_code, Bfrag);

    code_wmma_kernel<<<(N + 15) / 16, 256, 0, stream>>>(
        v_ws, Bfrag, nbr_idx, nbr_mask, b_code, attn_ws, res_ws, outp, N);
}
